// hyperConv_29454885716386
// MI455X (gfx1250) — matrix operand order repaired and checked by reading, not compiled, not run on hardware
//
#include <hip/hip_runtime.h>

#define B_    8
#define STYLE 128
#define CIN   256
#define COUT  256
#define HH    128
#define WW    128

typedef __attribute__((ext_vector_type(16))) __bf16 v16bf;
typedef __attribute__((ext_vector_type(8)))  __bf16 v8bf;
typedef __attribute__((ext_vector_type(8)))  float  v8f;

// ---------------------------------------------------------------------------
// Kernel 1: hyper-network heads.  f_k[b,9], f_cout[b,256], f_cin[b,256]
//   y = s @ W.T + b  (torch Linear convention)
// ---------------------------------------------------------------------------
__global__ __launch_bounds__(256) void hyper_heads_kernel(
    const float* __restrict__ s,
    const float* __restrict__ Wk,  const float* __restrict__ bk,
    const float* __restrict__ Wco, const float* __restrict__ bco,
    const float* __restrict__ Wci, const float* __restrict__ bci,
    float* __restrict__ fk, float* __restrict__ fco, float* __restrict__ fci)
{
    const int b = blockIdx.x;
    __shared__ float ssh[STYLE];
    for (int i = threadIdx.x; i < STYLE; i += blockDim.x) ssh[i] = s[b * STYLE + i];
    __syncthreads();

    const int NOUT = 9 + COUT + CIN;
    for (int j = threadIdx.x; j < NOUT; j += blockDim.x) {
        const float* wrow;
        float bias;
        float* dst;
        if (j < 9)              { wrow = Wk  + j * STYLE;            bias = bk[j];            dst = fk  + b * 9    + j; }
        else if (j < 9 + COUT)  { int o = j - 9;        wrow = Wco + o * STYLE; bias = bco[o]; dst = fco + b * COUT + o; }
        else                    { int c = j - 9 - COUT; wrow = Wci + c * STYLE; bias = bci[c]; dst = fci + b * CIN  + c; }
        float acc = bias;
        #pragma unroll 8
        for (int t = 0; t < STYLE; ++t) acc += ssh[t] * wrow[t];
        *dst = acc;
    }
}

// ---------------------------------------------------------------------------
// Kernel 2: per-sample modulated weights, swizzled for WMMA A-loads.
// Both rank-1 channel factors folded in (f_cin on c, f_k on tap p):
//   Wb[b][p][o][c] = bf16( W0[o][c][p] * f_k[b][p] * f_cin[b][c] )
// 9.4 MB bf16 total -> resident in the 192 MB L2 for the whole main kernel.
// ---------------------------------------------------------------------------
__global__ __launch_bounds__(256) void build_wb_kernel(
    const float* __restrict__ W0,     // [COUT,CIN,3,3]
    const float* __restrict__ fk,     // [B,9]
    const float* __restrict__ fci,    // [B,CIN]
    __bf16* __restrict__ Wb)          // [B,9,COUT,CIN]
{
    const size_t total = (size_t)B_ * 9 * COUT * CIN;
    for (size_t i = (size_t)blockIdx.x * blockDim.x + threadIdx.x; i < total;
         i += (size_t)gridDim.x * blockDim.x) {
        int    c = (int)(i % CIN);
        size_t t = i / CIN;
        int    o = (int)(t % COUT); t /= COUT;
        int    p = (int)(t % 9);
        int    b = (int)(t / 9);
        float w = W0[((size_t)o * CIN + c) * 9 + p];
        Wb[i] = (__bf16)(w * fk[b * 9 + p] * fci[b * CIN + c]);
    }
}

// ---------------------------------------------------------------------------
// Kernel 3: implicit-GEMM 3x3 conv via v_wmma_f32_16x16x32_bf16.
// Block: 256 thr = 8 waves.  One sample b, ALL 256 couts, one row y, 64 px.
//   wave -> 2 M-tiles (couts wave*16 and wave*16+128) x 4 N-tiles.
// So each input tile is staged exactly once per (b,y,wtile): HBM traffic
// = x once (134 MB) + out once (134 MB)  ->  bandwidth-optimal.
// K loop: CIN in chunks of 32; inner loop over the 9 taps.
// LDS: bf16 input tile, layout [col 0..65][row 0..2][ch 0..31].
// ---------------------------------------------------------------------------
#define WT     64          // output pixels per block
#define CH     32          // channel chunk (WMMA K)
#define COLS   (WT + 2)    // halo columns

__global__ __launch_bounds__(256) void hyper_conv_wmma_kernel(
    const float*  __restrict__ x,     // [B,CIN,H,W]
    const __bf16* __restrict__ Wb,    // [B,9,COUT,CIN]
    const float*  __restrict__ fco,   // [B,COUT]
    float*        __restrict__ out)   // [B,COUT,H,W]
{
    __shared__ alignas(16) __bf16 xs[COLS * 3 * CH];   // 12672 B

    const int wtile = blockIdx.x;            // 0..1
    const int y     = blockIdx.y;            // 0..127
    const int b     = blockIdx.z;            // 0..7
    const int xbase = wtile * WT;

    const int tid    = threadIdx.x;
    const int wave   = tid >> 5;             // 0..7
    const int lane   = tid & 31;
    const int laneM  = lane & 15;
    const int laneHi = lane >> 4;            // 0/1

    const int mrow0 = wave * 16 + laneM;     // A-row, M-tile 0 (couts 0..127)
    const int mrow1 = mrow0 + 128;           // A-row, M-tile 1 (couts 128..255)

    v8f acc[8];                              // [mt*4 + nt]
    #pragma unroll
    for (int a = 0; a < 8; ++a)
        #pragma unroll
        for (int r = 0; r < 8; ++r) acc[a][r] = 0.0f;

    const float*  xb  = x  + (size_t)b * CIN * HH * WW;
    const __bf16* Wbb = Wb + (size_t)b * 9 * COUT * CIN;

    for (int k0 = 0; k0 < CIN; k0 += CH) {
        __syncthreads();   // previous iteration's LDS reads done

        // ---- stage input tile (pure f32 -> bf16 convert; f_cin is in Wb) ----
        for (int idx = tid; idx < COLS * 3 * CH; idx += 256) {
            int col = idx % COLS;            // fastest: coalesced along w
            int t   = idx / COLS;
            int row = t % 3;
            int ch  = t / 3;
            int gx = xbase + col - 1;
            int gy = y + row - 1;
            float v = 0.0f;
            if ((unsigned)gx < (unsigned)WW && (unsigned)gy < (unsigned)HH)
                v = xb[(size_t)(k0 + ch) * HH * WW + (size_t)gy * WW + gx];
            xs[(col * 3 + row) * CH + ch] = (__bf16)v;
        }

        // ---- prefetch next K-chunk's input lines (global_prefetch_b8) ----
        if (k0 + CH < CIN && tid < 3 * CH) {
            int row = tid % 3;
            int ch  = tid / 3;
            int gy  = y + row - 1;
            if ((unsigned)gy < (unsigned)HH) {
                const float* pf = xb + (size_t)(k0 + CH + ch) * HH * WW
                                     + (size_t)gy * WW + xbase;
                __builtin_prefetch(pf, 0, 3);
                __builtin_prefetch(pf + 32, 0, 3);
            }
        }
        __syncthreads();

        // ---- 9 taps, WMMA accumulate, 2 M-tiles share each B fragment ----
        #pragma unroll
        for (int p = 0; p < 9; ++p) {
            const int dy = p / 3;
            const int dx = p % 3;

            // A fragments: Wbb[p][mrow][k0 + 8*laneHi (+0..7, +16..23)]
            const __bf16* a0p =
                Wbb + ((size_t)p * COUT + mrow0) * CIN + k0 + 8 * laneHi;
            const __bf16* a1p =
                Wbb + ((size_t)p * COUT + mrow1) * CIN + k0 + 8 * laneHi;
            v8bf a0lo = *reinterpret_cast<const v8bf*>(a0p);
            v8bf a0hi = *reinterpret_cast<const v8bf*>(a0p + 16);
            v8bf a1lo = *reinterpret_cast<const v8bf*>(a1p);
            v8bf a1hi = *reinterpret_cast<const v8bf*>(a1p + 16);
            v16bf af0, af1;
            #pragma unroll
            for (int i = 0; i < 8; ++i) {
                af0[i] = a0lo[i]; af0[8 + i] = a0hi[i];
                af1[i] = a1lo[i]; af1[8 + i] = a1hi[i];
            }

            #pragma unroll
            for (int nt = 0; nt < 4; ++nt) {
                // B fragment: column pixel = nt*16 + laneM (+dx halo shift)
                const int col = nt * 16 + laneM + dx;
                const __bf16* bptr = &xs[(col * 3 + dy) * CH + 8 * laneHi];
                v8bf blo = *reinterpret_cast<const v8bf*>(bptr);
                v8bf bhi = *reinterpret_cast<const v8bf*>(bptr + 16);
                v16bf bfrag;
                #pragma unroll
                for (int i = 0; i < 8; ++i) { bfrag[i] = blo[i]; bfrag[8 + i] = bhi[i]; }

                acc[nt] = __builtin_amdgcn_wmma_f32_16x16x32_bf16(
                    false, af0, false, bfrag, (short)0, acc[nt], false, false);
                acc[4 + nt] = __builtin_amdgcn_wmma_f32_16x16x32_bf16(
                    false, af1, false, bfrag, (short)0, acc[4 + nt], false, false);
            }
        }
    }

    // ---- epilogue: scale by f_cout, store f32 ----
    const float* fob = fco + b * COUT;
    #pragma unroll
    for (int mt = 0; mt < 2; ++mt) {
        #pragma unroll
        for (int nt = 0; nt < 4; ++nt) {
            const int xo = xbase + nt * 16 + laneM;
            #pragma unroll
            for (int r = 0; r < 8; ++r) {
                const int m = mt * 128 + wave * 16 + r + 8 * laneHi;
                out[(((size_t)b * COUT + m) * HH + y) * WW + xo] =
                    acc[mt * 4 + nt][r] * fob[m];
            }
        }
    }
}

// ---------------------------------------------------------------------------
// launch
// ---------------------------------------------------------------------------
extern "C" void kernel_launch(void* const* d_in, const int* in_sizes, int n_in,
                              void* d_out, int out_size, void* d_ws, size_t ws_size,
                              hipStream_t stream) {
    const float* x   = (const float*)d_in[0];   // [8,256,128,128]
    const float* s   = (const float*)d_in[1];   // [8,128]
    const float* W0  = (const float*)d_in[2];   // [256,256,3,3]
    const float* Wk  = (const float*)d_in[3];   // [9,128]
    const float* bk  = (const float*)d_in[4];   // [9]
    const float* Wco = (const float*)d_in[5];   // [256,128]
    const float* bco = (const float*)d_in[6];   // [256]
    const float* Wci = (const float*)d_in[7];   // [256,128]
    const float* bci = (const float*)d_in[8];   // [256]
    float* out = (float*)d_out;                 // [8,256,128,128]

    // workspace layout
    char* ws = (char*)d_ws;
    const size_t WB_BYTES = (size_t)B_ * 9 * COUT * CIN * sizeof(__bf16); // 9,437,184
    __bf16* Wb  = (__bf16*)(ws);
    float*  fk  = (float*)(ws + WB_BYTES);                 // 8*9
    float*  fco = (float*)(ws + WB_BYTES + 512);           // 8*256
    float*  fci = (float*)(ws + WB_BYTES + 512 + 8192);    // 8*256
    (void)in_sizes; (void)n_in; (void)out_size; (void)ws_size;

    hyper_heads_kernel<<<B_, 256, 0, stream>>>(s, Wk, bk, Wco, bco, Wci, bci,
                                               fk, fco, fci);
    build_wb_kernel<<<1024, 256, 0, stream>>>(W0, fk, fci, Wb);

    dim3 grid(WW / WT, HH, B_);                 // (2,128,8)
    hyper_conv_wmma_kernel<<<grid, 256, 0, stream>>>(x, Wb, fco, out);
}
